// AWDLSTMEncoder_21921513078948
// MI455X (gfx1250) — matrix operand order repaired and checked by reading, not compiled, not run on hardware
//
#include <hip/hip_runtime.h>
#include <hip/hip_bf16.h>

// ---------------------------------------------------------------------------
// AWD-LSTM encoder forward for MI455X (gfx1250, wave32, WMMA).
//   out = LSTM2(LSTM1(LSTM0(emb[x])))
// - bf16 data path, fp32 accumulation via v_wmma_f32_16x16x32_bf16.
// - Per layer: one large parallel GEMM precomputes xW = X*W_ih^T + b for all
//   T*B rows, then 512 per-step kernels run the recurrence (graph-captured;
//   kernel boundaries give the device-wide step barrier).
// - Step kernel stages the W_hh k-tile in LDS (double-buffered) via the
//   gfx1250 GLOBAL_LOAD_ASYNC_TO_LDS_B128 path (ASYNCcnt) so the fill
//   bypasses VGPRs and overlaps the WMMAs; reg round-trip fallback otherwise.
// - h double-buffered bf16 (no intra-step RAW race), c fp32.
// - 400-wide dims padded to 416 so K % 32 == 0 (pad lanes are zero).
// ---------------------------------------------------------------------------

typedef __bf16 bf16_t;
typedef bf16_t v16bf __attribute__((ext_vector_type(16)));
typedef bf16_t v8bf  __attribute__((ext_vector_type(8)));
typedef float  v8f   __attribute__((ext_vector_type(8)));
typedef int    v4i   __attribute__((ext_vector_type(4)));

#if defined(__has_builtin)
#if __has_builtin(__builtin_amdgcn_global_load_async_to_lds_b128) && \
    __has_builtin(__builtin_amdgcn_s_wait_asynccnt)
#define HAVE_ASYNC_LDS 1
#endif
#endif
#ifndef HAVE_ASYNC_LDS
#define HAVE_ASYNC_LDS 0
#endif

#if HAVE_ASYNC_LDS
// Builtin signature (from compiler diagnostic): global int4* src, LDS int4*
// dst, imm offset, imm cpol. AS1 prints as __device__, AS3 as __shared__.
typedef __attribute__((address_space(1))) v4i* gas_v4i;
typedef __attribute__((address_space(3))) v4i* las_v4i;
__device__ __forceinline__ void async_copy16(const void* gsrc, void* ldst) {
  // Generic->AS casts via integers: global generic addresses are value-
  // identical; LDS generic addresses truncate to the 32-bit LDS offset
  // (ISA 10.2 aperture rules).
  __builtin_amdgcn_global_load_async_to_lds_b128(
      (gas_v4i)(unsigned long long)gsrc,
      (las_v4i)(unsigned)(unsigned long long)ldst, 0, 0);
}
#endif

__device__ __forceinline__ v16bf frag_cat(v8bf lo, v8bf hi) {
  return __builtin_shufflevector(lo, hi, 0, 1, 2, 3, 4, 5, 6, 7,
                                         8, 9, 10, 11, 12, 13, 14, 15);
}

// A fragment: 16x32 (MxK) bf16, row-major, leading dim lda (elements).
// Layout: lanes 0-15 row M=lane, elems = K{0..15}; lanes 16-31
// hold K{16..31}. Two contiguous 16B loads per lane.
__device__ __forceinline__ v16bf load_a_frag(const bf16_t* __restrict__ A,
                                             int lda, int m0, int k0, int lane) {
  const bf16_t* r = A + (size_t)(m0 + (lane & 15)) * lda + k0 + 16 * (lane >> 4);
  v8bf lo = *(const v8bf*)(r);
  v8bf hi = *(const v8bf*)(r + 8);
  return frag_cat(lo, hi);
}

// B fragment: 32x16 (KxN), B[k][n] = W[n][k]; W row-major (N x K), ld ldw.
// Lane: col n = n0 + lane%16, K = k0 + 16*(lane/16) .. +16 (contiguous 32B).
__device__ __forceinline__ v16bf load_b_frag(const bf16_t* __restrict__ W,
                                             int ldw, int n0, int k0, int lane) {
  const bf16_t* r = W + (size_t)(n0 + (lane & 15)) * ldw + k0 + 16 * (lane >> 4);
  v8bf lo = *(const v8bf*)(r);
  v8bf hi = *(const v8bf*)(r + 8);
  return frag_cat(lo, hi);
}

__device__ __forceinline__ v8f wmma_bf16(v16bf a, v16bf b, v8f c) {
  return __builtin_amdgcn_wmma_f32_16x16x32_bf16(false, a, false, b,
                                                 (short)0, c, false, false);
}

__device__ __forceinline__ float sigm(float x) { return 1.0f / (1.0f + __expf(-x)); }

// ---------------------------------------------------------------------------
// Prologue kernels
// ---------------------------------------------------------------------------
__global__ void k_bias_sum(const float* __restrict__ a, const float* __restrict__ b,
                           float* __restrict__ o, int n) {
  int i = blockIdx.x * blockDim.x + threadIdx.x;
  if (i < n) o[i] = a[i] + b[i];
}

// fp32 (rows x cs) -> bf16 (rows x cd), zero-padding columns cs..cd-1
__global__ void k_cvt_pad(const float* __restrict__ src, bf16_t* __restrict__ dst,
                          int rows, int cs, int cd) {
  size_t i = (size_t)blockIdx.x * blockDim.x + threadIdx.x;
  size_t total = (size_t)rows * cd;
  if (i >= total) return;
  int c = (int)(i % cd);
  int r = (int)(i / cd);
  float v = (c < cs) ? src[(size_t)r * cs + c] : 0.0f;
  dst[i] = (bf16_t)v;
}

// Embedding gather -> bf16, padded to ld columns (cols 400..ld-1 zero)
__global__ void k_embed(const int* __restrict__ x, const float* __restrict__ emb,
                        bf16_t* __restrict__ X, int ld) {
  int r = blockIdx.x;                  // r = t*B + b
  int tok = x[r];
  const float* e = emb + (size_t)tok * 400;
  bf16_t* o = X + (size_t)r * ld;
  for (int c = threadIdx.x; c < ld; c += blockDim.x)
    o[c] = (bf16_t)((c < 400) ? e[c] : 0.0f);
}

__global__ void k_zero_state(bf16_t* __restrict__ h0, bf16_t* __restrict__ h1,
                             float* __restrict__ c, int n) {
  int i = blockIdx.x * blockDim.x + threadIdx.x;
  if (i < n) { h0[i] = (bf16_t)0.0f; h1[i] = (bf16_t)0.0f; c[i] = 0.0f; }
}

// ---------------------------------------------------------------------------
// Batched input projection: C[M,N] = A[M,K] * W[N,K]^T + bias[N]
// grid (N/64, M/128), 128 threads (4 waves). Wave w: 32x64 tile
// (rows 128*by + 32*w .. +32, cols 64*bx .. +64) -> 8 accumulators.
// 12 b128 loads per 8 WMMA (A-frag pair reused across 4 B tiles).
// ---------------------------------------------------------------------------
__global__ void __launch_bounds__(128)
k_gemm_xw(const bf16_t* __restrict__ A, int lda,
          const bf16_t* __restrict__ W, int ldw,
          const float* __restrict__ bias,
          float* __restrict__ C, int ldc, int K) {
  const int lane = threadIdx.x & 31;
  const int wave = threadIdx.x >> 5;
  const int m0 = blockIdx.y * 128 + wave * 32;
  const int n0 = blockIdx.x * 64;

  v8f acc[2][4] = {};
  for (int k0 = 0; k0 < K; k0 += 32) {
    if (k0 + 32 < K)  // prefetch next A k-tile (W stays hot in L2)
      __builtin_prefetch(A + (size_t)(m0 + (lane & 15)) * lda + k0 + 32, 0, 1);
    v16bf a0 = load_a_frag(A, lda, m0, k0, lane);
    v16bf a1 = load_a_frag(A, lda, m0 + 16, k0, lane);
#pragma unroll
    for (int j = 0; j < 4; ++j) {
      v16bf b = load_b_frag(W, ldw, n0 + 16 * j, k0, lane);
      acc[0][j] = wmma_bf16(a0, b, acc[0][j]);
      acc[1][j] = wmma_bf16(a1, b, acc[1][j]);
    }
  }

  const int half = lane >> 4;
  const int col = lane & 15;
#pragma unroll
  for (int mi = 0; mi < 2; ++mi) {
#pragma unroll
    for (int j = 0; j < 4; ++j) {
      int n = n0 + 16 * j + col;
      float bv = bias[n];
#pragma unroll
      for (int r = 0; r < 8; ++r) {
        int m = m0 + 16 * mi + r + 8 * half;
        C[(size_t)m * ldc + n] = acc[mi][j][r] + bv;
      }
    }
  }
}

// ---------------------------------------------------------------------------
// One LSTM time step: gates = xw_t + h_in * W_hh^T, fused cell update.
// grid = H/16 blocks, 128 threads (4 waves); wave w owns batch rows
// 16w..16w+15 and the block's 16 j-columns for ALL FOUR gates -> 4 accums.
// The 64x32 W_hh k-tile (4 gates x 16 j) is cooperatively staged into LDS
// (double-buffered, rows padded to 80B to spread banks) so each weight is
// read from L2 once per block instead of once per wave. Staging uses the
// gfx1250 async-to-LDS DMA (ASYNCcnt) when the builtin is available.
// ---------------------------------------------------------------------------
#define SROW 40  // LDS row stride in bf16 elements (80B, 16B-aligned)

__global__ void __launch_bounds__(128)
k_lstm_step(const float* __restrict__ xw,        // (64, 4H) slice at time t
            const bf16_t* __restrict__ h_in,     // (64, ldh) bf16
            bf16_t* __restrict__ h_out,          // (64, ldh) bf16
            float* __restrict__ c,               // (64, ldh) fp32
            int ldh,
            const bf16_t* __restrict__ Whh, int ldw,  // (4H, ldw) bf16
            int H, int K,
            bf16_t* __restrict__ out_bf, int ldo,     // next-layer input (or null)
            float* __restrict__ out_f) {              // final fp32 output (or null)
  __shared__ bf16_t smem[2][64 * SROW];

  const int tid = threadIdx.x;
  const int lane = tid & 31;
  const int wave = tid >> 5;
  const int m0 = wave * 16;              // batch rows
  const int j0 = blockIdx.x * 16;        // hidden columns

  // Stage one 64-row x 32-K weight tile: 256 16B chunks over 128 threads.
  auto stage = [&](int buf, int k0) {
#pragma unroll
    for (int i = 0; i < 2; ++i) {
      int ch = tid + i * 128;            // 0..255
      int row = ch >> 2;                 // 0..63  (= gate*16 + jj)
      int q = ch & 3;                    // 16B quarter within the 64B row
      int g = row >> 4, jj = row & 15;
      const bf16_t* src = Whh + (size_t)(g * H + j0 + jj) * ldw + k0 + q * 8;
      bf16_t* dst = &smem[buf][row * SROW + q * 8];
#if HAVE_ASYNC_LDS
      async_copy16(src, dst);
#else
      *(v8bf*)dst = *(const v8bf*)src;
#endif
    }
  };

  v8f acc[4] = {};
  const int KT = K >> 5;
  stage(0, 0);
#if HAVE_ASYNC_LDS
  __builtin_amdgcn_s_wait_asynccnt(0);
#endif
  __syncthreads();
  for (int kt = 0; kt < KT; ++kt) {
    const int cur = kt & 1;
    if (kt + 1 < KT) stage(cur ^ 1, (kt + 1) << 5);   // overlap with WMMA
    v16bf a = load_a_frag(h_in, ldh, m0, kt << 5, lane);
#pragma unroll
    for (int g = 0; g < 4; ++g) {        // gate order: i, f, g, o
      const bf16_t* s =
          &smem[cur][(g * 16 + (lane & 15)) * SROW + 16 * (lane >> 4)];
      v16bf b = frag_cat(*(const v8bf*)s, *(const v8bf*)(s + 8));
      acc[g] = wmma_bf16(a, b, acc[g]);
    }
#if HAVE_ASYNC_LDS
    __builtin_amdgcn_s_wait_asynccnt(0);  // next buffer landed in LDS
#endif
    __syncthreads();  // reads of `cur` done; stores to `cur^1` visible
  }

  const int half = lane >> 4;
  const int j = j0 + (lane & 15);
  const int n4 = 4 * H;
#pragma unroll
  for (int r = 0; r < 8; ++r) {
    int b = m0 + r + 8 * half;
    const float* xwb = xw + (size_t)b * n4;
    float gi = acc[0][r] + xwb[j];
    float gf = acc[1][r] + xwb[H + j];
    float gg = acc[2][r] + xwb[2 * H + j];
    float go = acc[3][r] + xwb[3 * H + j];
    float cp = c[(size_t)b * ldh + j];
    float cn = sigm(gf) * cp + sigm(gi) * tanhf(gg);
    float hn = sigm(go) * tanhf(cn);
    c[(size_t)b * ldh + j] = cn;
    h_out[(size_t)b * ldh + j] = (bf16_t)hn;
    if (out_bf) out_bf[(size_t)b * ldo + j] = (bf16_t)hn;
    if (out_f)  out_f[(size_t)b * 400 + j] = hn;
  }
}

// ---------------------------------------------------------------------------
// Host orchestration (all launches on `stream`; graph-capture safe)
// ---------------------------------------------------------------------------
extern "C" void kernel_launch(void* const* d_in, const int* in_sizes, int n_in,
                              void* d_out, int out_size, void* d_ws, size_t ws_size,
                              hipStream_t stream) {
  (void)in_sizes; (void)n_in; (void)out_size; (void)ws_size;
  const int T = 512, B = 64, R = T * B;

  const int*   x   = (const int*)d_in[0];
  const float* emb = (const float*)d_in[1];
  const float* Wih[3] = { (const float*)d_in[2], (const float*)d_in[6], (const float*)d_in[10] };
  const float* Whh[3] = { (const float*)d_in[3], (const float*)d_in[7], (const float*)d_in[11] };
  const float* bih[3] = { (const float*)d_in[4], (const float*)d_in[8], (const float*)d_in[12] };
  const float* bhh[3] = { (const float*)d_in[5], (const float*)d_in[9], (const float*)d_in[13] };

  // Layer shapes (din -> H); 400-dims padded to 416 for K%32==0.
  const int din_src[3] = { 400, 1152, 1152 };
  const int din_pad[3] = { 416, 1152, 1152 };
  const int hdim[3]    = { 1152, 1152, 400 };
  const int hpad[3]    = { 1152, 1152, 416 };
  const int N4[3]      = { 4608, 4608, 1600 };

  // Carve workspace (256B-aligned regions).
  char* p = (char*)d_ws;
  auto carve = [&](size_t bytes) -> void* {
    char* q = p; p += (bytes + 255) & ~(size_t)255; return (void*)q;
  };
  bf16_t* Xa  = (bf16_t*)carve((size_t)R * 1152 * sizeof(bf16_t));
  bf16_t* Xb  = (bf16_t*)carve((size_t)R * 1152 * sizeof(bf16_t));
  float*  xW  = (float*) carve((size_t)R * 4608 * sizeof(float));
  bf16_t* wih_b[3]; bf16_t* whh_b[3]; float* bias_b[3];
  for (int l = 0; l < 3; ++l) {
    wih_b[l]  = (bf16_t*)carve((size_t)N4[l] * din_pad[l] * sizeof(bf16_t));
    whh_b[l]  = (bf16_t*)carve((size_t)N4[l] * hpad[l]    * sizeof(bf16_t));
    bias_b[l] = (float*) carve((size_t)N4[l] * sizeof(float));
  }
  bf16_t* h0 = (bf16_t*)carve((size_t)B * 1152 * sizeof(bf16_t));
  bf16_t* h1 = (bf16_t*)carve((size_t)B * 1152 * sizeof(bf16_t));
  float*  cb = (float*) carve((size_t)B * 1152 * sizeof(float));

  // --- Prologue: bias sums, padded bf16 weight conversion, embedding ---
  for (int l = 0; l < 3; ++l) {
    k_bias_sum<<<(N4[l] + 255) / 256, 256, 0, stream>>>(bih[l], bhh[l], bias_b[l], N4[l]);
    size_t ti = (size_t)N4[l] * din_pad[l];
    k_cvt_pad<<<(unsigned)((ti + 255) / 256), 256, 0, stream>>>(
        Wih[l], wih_b[l], N4[l], din_src[l], din_pad[l]);
    size_t th = (size_t)N4[l] * hpad[l];
    k_cvt_pad<<<(unsigned)((th + 255) / 256), 256, 0, stream>>>(
        Whh[l], whh_b[l], N4[l], hdim[l], hpad[l]);
  }
  k_embed<<<R, 128, 0, stream>>>(x, emb, Xa, 416);

  // --- Layers ---
  bf16_t* layer_in[3]  = { Xa, Xb, Xa };
  bf16_t* layer_out[3] = { Xb, Xa, nullptr };

  for (int l = 0; l < 3; ++l) {
    const int N = N4[l], K = din_pad[l];

    // xW = X * W_ih^T + (b_ih + b_hh), for all T*B rows at once.
    dim3 g(N / 64, R / 128);
    k_gemm_xw<<<g, 128, 0, stream>>>(layer_in[l], K, wih_b[l], K, bias_b[l], xW, N, K);

    // h0 = h1 = c = 0
    const int ns = B * 1152;
    k_zero_state<<<(ns + 255) / 256, 256, 0, stream>>>(h0, h1, cb, ns);

    const int H = hdim[l], Kh = hpad[l];
    const int jblocks = H / 16;
    for (int t = 0; t < T; ++t) {
      bf16_t* hin  = (t & 1) ? h1 : h0;
      bf16_t* hout = (t & 1) ? h0 : h1;
      const float* xwt = xW + (size_t)t * B * N;
      bf16_t* ob = layer_out[l] ? layer_out[l] + (size_t)t * B * 1152 : nullptr;
      float*  of = (l == 2) ? ((float*)d_out + (size_t)t * B * 400) : nullptr;
      k_lstm_step<<<jblocks, 128, 0, stream>>>(
          xwt, hin, hout, cb, Kh, whh_b[l], Kh, H, Kh, ob, 1152, of);
    }
  }
}
